// GAT_Model_59399397703791
// MI455X (gfx1250) — matrix operand order repaired and checked by reading, not compiled, not run on hardware
//
#include <hip/hip_runtime.h>
#include <hip/hip_bf16.h>

// GAT (3-layer GATConv w/ edge attrs) + global_add_pool + MLP head, MI455X gfx1250.
// Node GEMM on the WMMA pipe: v_wmma_f32_16x16x32_bf16, f32 accumulate.
// A tiles staged in LDS in fragment order (ds_load_b128 per operand half);
// W pre-packed per layer into global fragment order (coalesced b128 loads, L2-hot).
// Edge MLP algebraically folded into a 16x4 matrix (el only feeds a per-head dot).

typedef __attribute__((ext_vector_type(16))) __bf16 v16bf;
typedef __attribute__((ext_vector_type(8)))  float  v8f;

#define F_IN   64
#define HID    256
#define HEADS  4
#define CPH    64      // channels per head
#define EDIM   16
#define NEG_SLOPE 0.2f

// ---------- helpers ----------
__device__ __forceinline__ unsigned fkey(float f) {
  unsigned u = __float_as_uint(f);
  return (u & 0x80000000u) ? ~u : (u | 0x80000000u);   // monotonic float->uint
}
__device__ __forceinline__ float funkey(unsigned k) {
  return (k & 0x80000000u) ? __uint_as_float(k & 0x7FFFFFFFu) : __uint_as_float(~k);
}

__global__ void zero_f32(float* __restrict__ p, int n) {
  int i = blockIdx.x * blockDim.x + threadIdx.x;
  if (i < n) p[i] = 0.0f;
}

// Pack W [K,256] f32 -> bf16 in WMMA B-fragment order:
//   Wp[((ch*16 + coltile)*32 + lane)*16 + elem]
//   lane = (col&15) + (K16-half ? 16 : 0), elem = k&15, ch = k>>5, coltile = col>>4
__global__ void pack_w(const float* __restrict__ W, __bf16* __restrict__ Wp, int K) {
  int i = blockIdx.x * blockDim.x + threadIdx.x;
  if (i >= K * HID) return;
  int k = i >> 8, c = i & (HID - 1);
  int ch = k >> 5, kk = k & 31;
  int ct = c >> 4;
  int ln = (c & 15) + ((kk & 16) ? 16 : 0);
  int el = kk & 15;
  Wp[((ch * 16 + ct) * 32 + ln) * 16 + el] = (__bf16)W[i];
}

// deg + sum of incoming edge attrs (for self-loop fill_value='mean')
__global__ void deg_loopsum(const int* __restrict__ dst, const float* __restrict__ ea,
                            float* __restrict__ deg, float* __restrict__ loopsum, int E) {
  int i = blockIdx.x * blockDim.x + threadIdx.x;
  if (i >= E * EDIM) return;
  int e = i >> 4, k = i & 15;
  int d = dst[e];
  atomicAdd(&loopsum[d * EDIM + k], ea[i]);
  if (k == 0) atomicAdd(&deg[d], 1.0f);
}

__global__ void loop_div(float* __restrict__ loopattr, const float* __restrict__ deg, int N) {
  int i = blockIdx.x * blockDim.x + threadIdx.x;
  if (i >= N * EDIM) return;
  loopattr[i] = loopattr[i] / fmaxf(deg[i >> 4], 1.0f);
}

// ---------- WMMA node GEMM: C[M,256] = A[M,K] @ W[K,256] ----------
// 256 threads = 8 waves; wave w owns 16x16 tile at col tile (blockIdx.y*8 + w).
// A slab (16xK) staged once in LDS in fragment order; B fragments read straight
// from the pre-packed Wp (contiguous 32B per lane, coalesced across lanes).
template <int K>
__global__ void gemm_wmma(const float* __restrict__ A, const __bf16* __restrict__ Wp,
                          float* __restrict__ C, int M) {
  constexpr int NCH = K / 32;
  const int tid  = threadIdx.x;
  const int wave = tid >> 5;
  const int lane = tid & 31;
  const int row0 = blockIdx.x * 16;
  const int ctg  = blockIdx.y * 8 + wave;   // global col tile 0..15

  __shared__ __bf16 As[NCH][32][16];        // <= 8 KB, fragment order

  // stage A slab (16 x K f32 -> bf16 frag layout), coalesced global reads
  #pragma unroll
  for (int j = 0; j < (16 * K) / 256; ++j) {
    int i = tid + j * 256;
    int r = i / K, c = i % K;               // K is power of two -> shifts
    int gr = row0 + r; if (gr >= M) gr = M - 1;
    float v = A[gr * K + c];
    int ch = c >> 5, cc = c & 31;
    int ln = r + ((cc & 16) ? 16 : 0);
    int el = cc & 15;
    As[ch][ln][el] = (__bf16)v;
  }
  __syncthreads();

  v8f acc = {};
  #pragma unroll
  for (int ch = 0; ch < NCH; ++ch) {
    v16bf af = *(const v16bf*)(&As[ch][lane][0]);                       // ds_load_b128 x2
    v16bf bf = *(const v16bf*)(Wp + ((ch * 16 + ctg) * 32 + lane) * 16); // global b128 x2
    acc = __builtin_amdgcn_wmma_f32_16x16x32_bf16(false, af, false, bf, (short)0, acc,
                                                  false, false);
  }

  // C/D layout: VGPR i -> (lane<16 ? M=i : M=8+i), N = lane&15
  int col  = ctg * 16 + (lane & 15);
  int rofs = (lane < 16) ? 0 : 8;
  #pragma unroll
  for (int i = 0; i < 8; ++i) {
    int r = row0 + rofs + i;
    if (r < M) C[r * HID + col] = acc[i];
  }
}

// per-node attention logits: alpha_src/dst[n,h] = <xl[n,h,:], a_src/dst[h,:]>
__global__ void alpha_node(const float* __restrict__ xl, const float* __restrict__ a_src,
                           const float* __restrict__ a_dst, float* __restrict__ asrc,
                           float* __restrict__ adst, int N) {
  int i = blockIdx.x * blockDim.x + threadIdx.x;
  if (i >= N * HEADS) return;
  int n = i >> 2, h = i & 3;
  const float* xr = xl + n * HID + h * CPH;
  float ss = 0.f, sd = 0.f;
  #pragma unroll 4
  for (int c = 0; c < CPH; ++c) {
    float v = xr[c];
    ss += v * a_src[h * CPH + c];
    sd += v * a_dst[h * CPH + c];
  }
  asrc[i] = ss; adst[i] = sd;
}

// fold edge MLP: M[k,h] = sum_c We[k, h*64+c] * a_edge[h,c]   (16x4)
__global__ void fold_edge_mat(const float* __restrict__ We, const float* __restrict__ a_edge,
                              float* __restrict__ Medge) {
  int i = threadIdx.x;            // 64 threads
  int k = i >> 2, h = i & 3;
  float s = 0.f;
  for (int c = 0; c < CPH; ++c) s += We[k * HID + h * CPH + c] * a_edge[h * CPH + c];
  Medge[i] = s;
}

// alpha = leaky_relu(asrc[s] + adst[d] + ea_full @ M); segment_max via uint-key atomicMax
__global__ void alpha_edge_max(const int* __restrict__ src, const int* __restrict__ dst,
                               const float* __restrict__ ea, const float* __restrict__ loopattr,
                               const float* __restrict__ Medge, const float* __restrict__ asrc,
                               const float* __restrict__ adst, float* __restrict__ alpha,
                               unsigned* __restrict__ amaxk, int E, int N) {
  int i = blockIdx.x * blockDim.x + threadIdx.x;
  if (i >= (E + N) * HEADS) return;
  int e = i >> 2, h = i & 3;
  int s, d; const float* er;
  if (e < E) { s = src[e]; d = dst[e]; er = ea + e * EDIM; }
  else       { s = e - E;  d = e - E;  er = loopattr + (e - E) * EDIM; }
  float aed = 0.f;
  #pragma unroll
  for (int k = 0; k < EDIM; ++k) aed += er[k] * Medge[k * HEADS + h];
  float a = asrc[s * HEADS + h] + adst[d * HEADS + h] + aed;
  a = (a > 0.f) ? a : NEG_SLOPE * a;
  alpha[i] = a;
  atomicMax(&amaxk[d * HEADS + h], fkey(a));
}

__global__ void exp_den(const int* __restrict__ dst, float* __restrict__ alpha,
                        const unsigned* __restrict__ amaxk, float* __restrict__ den,
                        int E, int N) {
  int i = blockIdx.x * blockDim.x + threadIdx.x;
  if (i >= (E + N) * HEADS) return;
  int e = i >> 2, h = i & 3;
  int d = (e < E) ? dst[e] : (e - E);
  float ex = __expf(alpha[i] - funkey(amaxk[d * HEADS + h]));
  alpha[i] = ex;
  atomicAdd(&den[d * HEADS + h], ex);
}

// one block per edge; lane c moves one channel: hout[d,c] += coef[h(c)] * xl[s,c]
__global__ void scatter_msg(const int* __restrict__ src, const int* __restrict__ dst,
                            const float* __restrict__ ex, const float* __restrict__ den,
                            const float* __restrict__ xl, float* __restrict__ hout, int E) {
  int e = blockIdx.x;
  int c = threadIdx.x;          // 0..255
  int h = c >> 6;
  int s, d;
  if (e < E) { s = src[e]; d = dst[e]; }
  else       { s = e - E;  d = e - E; }
  float coef = ex[e * HEADS + h] / (den[d * HEADS + h] + 1e-16f);
  atomicAdd(&hout[d * HID + c], coef * xl[s * HID + c]);
}

__global__ void bias_act(float* __restrict__ h, const float* __restrict__ b, int N, int relu) {
  int i = blockIdx.x * blockDim.x + threadIdx.x;
  if (i >= N * HID) return;
  float v = h[i] + b[i & (HID - 1)];
  h[i] = relu ? fmaxf(v, 0.f) : v;
}

__global__ void pool_add(const float* __restrict__ h, const int* __restrict__ batch,
                         float* __restrict__ g, int N) {
  int i = blockIdx.x * blockDim.x + threadIdx.x;
  if (i >= N * HID) return;
  int n = i >> 8, c = i & (HID - 1);
  atomicAdd(&g[batch[n] * HID + c], h[i]);
}

// per-graph MLP head: relu(g @ W1 + b1) @ W2 + b2 ; 64 threads/block
__global__ void mlp_head(const float* __restrict__ g, const float* __restrict__ w1,
                         const float* __restrict__ b1, const float* __restrict__ w2,
                         const float* __restrict__ b2, float* __restrict__ out) {
  int gi = blockIdx.x, j = threadIdx.x;     // j in [0,64)
  const float* gr = g + gi * HID;
  float t = b1[j];
  #pragma unroll 4
  for (int k = 0; k < HID; ++k) t += gr[k] * w1[k * 64 + j];
  t = fmaxf(t, 0.f) * w2[j];
  __shared__ float red[64];
  red[j] = t; __syncthreads();
  for (int s2 = 32; s2 > 0; s2 >>= 1) { if (j < s2) red[j] += red[j + s2]; __syncthreads(); }
  if (j == 0) out[gi] = red[0] + b2[0];
}

// ---------------------------------------------------------------------------
extern "C" void kernel_launch(void* const* d_in, const int* in_sizes, int n_in,
                              void* d_out, int out_size, void* d_ws, size_t ws_size,
                              hipStream_t stream) {
  const int N = in_sizes[0] / F_IN;      // 100000
  const int E = in_sizes[2] / EDIM;      // 300000
  const int G = out_size;                // 4096
  const int ET = E + N;                  // edges incl. self-loops

  const float* x       = (const float*)d_in[0];
  const int*   eidx    = (const int*)  d_in[1];
  const float* ea      = (const float*)d_in[2];
  const int*   batch   = (const int*)  d_in[3];
  // lists flatten in order: Ws[0..2], att_src[0..2], att_dst[0..2], We[0..2], att_edge[0..2], biases[0..2]
  const float* Ws[3]      = {(const float*)d_in[4],  (const float*)d_in[5],  (const float*)d_in[6]};
  const float* a_src[3]   = {(const float*)d_in[7],  (const float*)d_in[8],  (const float*)d_in[9]};
  const float* a_dst[3]   = {(const float*)d_in[10], (const float*)d_in[11], (const float*)d_in[12]};
  const float* We[3]      = {(const float*)d_in[13], (const float*)d_in[14], (const float*)d_in[15]};
  const float* a_edge[3]  = {(const float*)d_in[16], (const float*)d_in[17], (const float*)d_in[18]};
  const float* bias[3]    = {(const float*)d_in[19], (const float*)d_in[20], (const float*)d_in[21]};
  const float* lin1_w = (const float*)d_in[22];
  const float* lin1_b = (const float*)d_in[23];
  const float* lin2_w = (const float*)d_in[24];
  const float* lin2_b = (const float*)d_in[25];
  float* out = (float*)d_out;

  const int* src = eidx;
  const int* dst = eidx + E;

  // workspace carve (256B aligned)
  char* base = (char*)d_ws;
  size_t off = 0;
  auto carve = [&](size_t bytes) -> void* {
    void* p = base + off;
    off = (off + bytes + 255) & ~(size_t)255;
    return p;
  };
  float*    h1       = (float*)   carve((size_t)N * HID * 4);
  float*    h2       = (float*)   carve((size_t)N * HID * 4);
  float*    xl       = (float*)   carve((size_t)N * HID * 4);
  float*    alpha    = (float*)   carve((size_t)ET * HEADS * 4);
  unsigned* amaxk    = (unsigned*)carve((size_t)N * HEADS * 4);
  float*    den      = (float*)   carve((size_t)N * HEADS * 4);
  float*    asrc     = (float*)   carve((size_t)N * HEADS * 4);
  float*    adst     = (float*)   carve((size_t)N * HEADS * 4);
  float*    deg      = (float*)   carve((size_t)N * 4);
  float*    loopattr = (float*)   carve((size_t)N * EDIM * 4);
  float*    Medge    = (float*)   carve(EDIM * HEADS * 4);
  __bf16*   Wp       = (__bf16*)  carve((size_t)HID * HID * 2);  // packed W (frag order)
  float*    pooled   = (float*)   carve((size_t)G * HID * 4);
  (void)ws_size; (void)n_in;

  const int T = 256;
  auto blk = [&](long long n) { return (unsigned)((n + T - 1) / T); };

  // self-loop attrs: mean of incoming edge attrs (computed once; identical across layers)
  zero_f32<<<blk(N), T, 0, stream>>>(deg, N);
  zero_f32<<<blk((long long)N * EDIM), T, 0, stream>>>(loopattr, N * EDIM);
  deg_loopsum<<<blk((long long)E * EDIM), T, 0, stream>>>(dst, ea, deg, loopattr, E);
  loop_div<<<blk((long long)N * EDIM), T, 0, stream>>>(loopattr, deg, N);

  for (int l = 0; l < 3; ++l) {
    const float* hin  = (l == 0) ? x : ((l == 1) ? h1 : h2);
    float*       hout = (l == 1) ? h2 : h1;
    const int    K    = (l == 0) ? F_IN : HID;

    zero_f32<<<blk(N * HEADS), T, 0, stream>>>((float*)amaxk, N * HEADS);
    zero_f32<<<blk(N * HEADS), T, 0, stream>>>(den, N * HEADS);
    zero_f32<<<blk((long long)N * HID), T, 0, stream>>>(hout, N * HID);

    // node GEMM on the WMMA pipe (bf16 in / f32 accum), W pre-packed to frag order
    pack_w<<<blk((long long)K * HID), T, 0, stream>>>(Ws[l], Wp, K);
    if (K == F_IN)
      gemm_wmma<F_IN><<<dim3((N + 15) / 16, 2), 256, 0, stream>>>(hin, Wp, xl, N);
    else
      gemm_wmma<HID><<<dim3((N + 15) / 16, 2), 256, 0, stream>>>(hin, Wp, xl, N);

    alpha_node<<<blk((long long)N * HEADS), T, 0, stream>>>(xl, a_src[l], a_dst[l], asrc, adst, N);
    fold_edge_mat<<<1, 64, 0, stream>>>(We[l], a_edge[l], Medge);

    alpha_edge_max<<<blk((long long)ET * HEADS), T, 0, stream>>>(
        src, dst, ea, loopattr, Medge, asrc, adst, alpha, amaxk, E, N);
    exp_den<<<blk((long long)ET * HEADS), T, 0, stream>>>(dst, alpha, amaxk, den, E, N);
    scatter_msg<<<ET, 256, 0, stream>>>(src, dst, alpha, den, xl, hout, E);
    bias_act<<<blk((long long)N * HID), T, 0, stream>>>(hout, bias[l], N, (l < 2) ? 1 : 0);
  }

  // global_add_pool + MLP head (h1 holds layer-2 output)
  zero_f32<<<blk((long long)G * HID), T, 0, stream>>>(pooled, G * HID);
  pool_add<<<blk((long long)N * HID), T, 0, stream>>>(h1, batch, pooled, N);
  mlp_head<<<G, 64, 0, stream>>>(pooled, lin1_w, lin1_b, lin2_w, lin2_b, out);
}
